// LinearLatent_58935541236353
// MI455X (gfx1250) — matrix operand order repaired and checked by reading, not compiled, not run on hardware
//
#include <hip/hip_runtime.h>

// ---------------------------------------------------------------------------
// LinearLatent pipeline for MI455X (gfx1250, wave32, WMMA)
//   out[b,n,s] = sum_l (latent0 @ T^(s+1))[b,l] * transform[n,l]
// Decomposition: s = 16*j + i
//   K_cvt   : transform -> bf16
//   K_gemm  : latent0 = x @ transform (fp32); T^2,T^4,T^8,T^16 (fp32)
//   K_rchain: R[j] = latent0 @ (T^16)^j   (fp32, row-parallel)
//   K_lats  : lats[16j+i] = R[j] @ T^(i+1) via V_WMMA_F32_16X16X4_F32 (fp32),
//             stored bf16 into latT[b][j][l][i] (contiguous WMMA B-panels)
//   K_proj  : out[b, n, 16j+i] via V_WMMA_F32_16X16X32_BF16, NT stores
//             (268 MB output stream kept out of L2 so latT/transform stay hot)
// ---------------------------------------------------------------------------

typedef __bf16 bf16;
typedef __attribute__((ext_vector_type(8)))  __bf16       v8bf;
typedef __attribute__((ext_vector_type(16))) __bf16       v16bf;
typedef __attribute__((ext_vector_type(8)))  float        v8f;
typedef __attribute__((ext_vector_type(2)))  float        v2f;
typedef __attribute__((ext_vector_type(4)))  unsigned int v4u;

#define BATCH 256
#define NIN   1024
#define NLAT  128
#define STEPS 256   // 16 chunks * 16 inner steps

static __device__ __forceinline__ unsigned short f32_to_bf16_bits(float f) {
  __bf16 h = (__bf16)f;
  return __builtin_bit_cast(unsigned short, h);
}

// --------------------------- trivial converters ----------------------------
__global__ void k_cvt_bf16(const float* __restrict__ src,
                           bf16* __restrict__ dst, int n) {
  int id = blockIdx.x * blockDim.x + threadIdx.x;
  if (id < n) dst[id] = (bf16)src[id];
}

// --------------------- generic small fp32 GEMM (VALU) ----------------------
// C[M,N] = A[M,K] @ B[K,N]; one thread per output element. Used only for the
// tiny setup GEMMs (latent0 and the 128^3 squarings) -- sub-µs each.
__global__ void k_gemm_f32(const float* __restrict__ A,
                           const float* __restrict__ B,
                           float* __restrict__ C, int M, int N, int K) {
  int id = blockIdx.x * blockDim.x + threadIdx.x;
  if (id >= M * N) return;
  int m = id / N, n = id % N;
  float acc = 0.f;
#pragma unroll 4
  for (int k = 0; k < K; ++k)
    acc += __builtin_nontemporal_load(&A[(size_t)m * K + k]) *
           B[(size_t)k * N + n];
  C[id] = acc;
}

// ------------------- R-chain: R[j] = latent0 @ (T^16)^j --------------------
// Rows of the recurrence are independent -> 64 WGs x 4 rows, 15 internal
// steps, no cross-WG sync. fp32 throughout (precision-critical chain).
__global__ __launch_bounds__(256) void k_rchain(
    const float* __restrict__ lat0,   // [256][128]
    const float* __restrict__ T16,    // [128][128]
    float* __restrict__ R) {          // [16][256][128]
  __shared__ float Rl[4 * NLAT];
  const int g = blockIdx.x;           // 0..63
  const int tid = threadIdx.x;
  const int b0 = g * 4;

  for (int t = tid; t < 4 * NLAT; t += 256) Rl[t] = lat0[(size_t)b0 * NLAT + t];
  __syncthreads();
  for (int t = tid; t < 4 * NLAT; t += 256) R[(size_t)b0 * NLAT + t] = Rl[t];

  for (int jj = 1; jj < 16; ++jj) {
    const int p0 = tid, p1 = tid + 256;        // (row,col) output pairs
    const int bb0 = p0 >> 7, l0 = p0 & 127;
    const int bb1 = p1 >> 7, l1 = p1 & 127;
    float a0 = 0.f, a1 = 0.f;
#pragma unroll 4
    for (int k = 0; k < NLAT; ++k) {
      const float r0 = Rl[bb0 * NLAT + k];
      const float r1 = Rl[bb1 * NLAT + k];
      a0 += r0 * T16[k * NLAT + l0];           // coalesced, L2-hot
      a1 += r1 * T16[k * NLAT + l1];
    }
    __syncthreads();
    Rl[p0] = a0; Rl[p1] = a1;
    __syncthreads();
    R[((size_t)jj * BATCH + b0) * NLAT + p0] = a0;
    R[((size_t)jj * BATCH + b0) * NLAT + p1] = a1;
  }
}

// ---------------- K_lats: 16 chunk-local steps, fp32 WMMA ------------------
// WG = (b-tile of 16, chunk j). G_0 = R[j] tile; G_{i+1} = G_i @ T.
// G kept transposed in LDS ([k][m], rows padded to 17 dwords: 17 coprime 64
// -> conflict-free strided writes). T read from global (64 KB, WGP$/L2-hot).
// Each step's G is rounded to bf16 into per-lane registers and written out
// once at the end as contiguous 32B runs of latT[b][j][l][0..15].
__global__ __launch_bounds__(256) void k_lats(
    const float* __restrict__ T,      // [128][128] transition fp32
    const float* __restrict__ R,      // [16][256][128]
    bf16* __restrict__ latT) {        // [256][16][128][16]
  __shared__ float GT[NLAT * 17];

  const int wg = blockIdx.x;          // 0..255
  const int bt = wg >> 4;             // b-tile
  const int j  = wg & 15;             // chunk
  const int tid = threadIdx.x;

  for (int t = tid; t < 16 * NLAT; t += 256) {
    int m = t >> 7, k = t & 127;
    GT[k * 17 + m] = R[((size_t)j * BATCH + bt * 16 + m) * NLAT + k];
  }
  __syncthreads();

#if __has_builtin(__builtin_amdgcn_wmma_f32_16x16x4_f32)
  const int lane = tid & 31;
  const int mn   = lane & 15;         // A row (b) == B col (l) == D col
  const int half = lane >> 4;
  const int lt   = tid >> 5;          // wave id = l-tile 0..7
  const int lcol = lt * 16 + mn;

  unsigned short gb[8][16];           // [D-row r][step i] bf16 bits

#pragma unroll
  for (int i = 0; i < 16; ++i) {
    v8f d = {};
#pragma unroll 4
    for (int kc = 0; kc < 32; ++kc) {
      const int ka = kc * 4 + 2 * half;       // K pair per ISA f32 layout
      v2f a, b;
      a.x = GT[ka * 17 + mn];
      a.y = GT[(ka + 1) * 17 + mn];
      b.x = T[ka * NLAT + lcol];
      b.y = T[(ka + 1) * NLAT + lcol];
      d = __builtin_amdgcn_wmma_f32_16x16x4_f32(false, a, false, b,
                                                (short)0, d, false, false);
    }
    __syncthreads();                  // all reads of GT done
#pragma unroll
    for (int r = 0; r < 8; ++r) {
      const int bb = r + 8 * half;    // D row = b index
      GT[lcol * 17 + bb] = d[r];      // new G, transposed for next step
      gb[r][i] = f32_to_bf16_bits(d[r]);
    }
    __syncthreads();
  }

#pragma unroll
  for (int r = 0; r < 8; ++r) {
    const int bb = r + 8 * half;
    bf16* dst = latT + ((((size_t)(bt * 16 + bb) * 16 + j) * NLAT + lcol) * 16);
    unsigned int u[8];
#pragma unroll
    for (int e = 0; e < 8; ++e)
      u[e] = (unsigned)gb[r][2 * e] | ((unsigned)gb[r][2 * e + 1] << 16);
    v4u w0 = {u[0], u[1], u[2], u[3]};
    v4u w1 = {u[4], u[5], u[6], u[7]};
    *(v4u*)dst = w0;                  // 32B contiguous per (b,l)
    *(((v4u*)dst) + 1) = w1;
  }
#else
  // VALU fp32 fallback (same math, thread-per-output)
  unsigned short gb[8][16];
#pragma unroll
  for (int i = 0; i < 16; ++i) {
    float acc[8];
#pragma unroll
    for (int q = 0; q < 8; ++q) {
      const int p = tid + 256 * q;
      const int mb = p >> 7, ll = p & 127;
      float a = 0.f;
#pragma unroll 4
      for (int k = 0; k < NLAT; ++k) a += GT[k * 17 + mb] * T[k * NLAT + ll];
      acc[q] = a;
    }
    __syncthreads();
#pragma unroll
    for (int q = 0; q < 8; ++q) {
      const int p = tid + 256 * q;
      const int mb = p >> 7, ll = p & 127;
      GT[ll * 17 + mb] = acc[q];
      gb[q][i] = f32_to_bf16_bits(acc[q]);
    }
    __syncthreads();
  }
#pragma unroll
  for (int q = 0; q < 8; ++q) {
    const int p = tid + 256 * q;
    const int mb = p >> 7, ll = p & 127;
    bf16* dst = latT + ((((size_t)(bt * 16 + mb) * 16 + j) * NLAT + ll) * 16);
    unsigned int u[8];
#pragma unroll
    for (int e = 0; e < 8; ++e)
      u[e] = (unsigned)gb[q][2 * e] | ((unsigned)gb[q][2 * e + 1] << 16);
    v4u w0 = {u[0], u[1], u[2], u[3]};
    v4u w1 = {u[4], u[5], u[6], u[7]};
    *(v4u*)dst = w0;
    *(((v4u*)dst) + 1) = w1;
  }
#endif
}

// ------------------ K_proj: the 268 MB memory-bound GEMM -------------------
// One wave per 16(n) x 16(s) output tile of one b. 4 x v_wmma_f32_16x16x32_bf16.
// A = transform bf16 (L2-resident, shared by all waves of a block),
// B = latT slab (contiguous 4KB per (b,j)). Output stores are non-temporal
// (TH=NT): the 268 MB write stream exceeds the 192 MB L2 and must not evict
// the reused latT/transform operands.
__global__ __launch_bounds__(256) void k_proj(
    const bf16* __restrict__ tfm,     // [1024][128] bf16
    const bf16* __restrict__ latT,    // [256][16][128][16]
    float* __restrict__ out) {        // [256][1024][256]
  const int w    = blockIdx.x * 8 + (threadIdx.x >> 5);
  const int lane = threadIdx.x & 31;
  const int j  = w & 15;
  const int nt = (w >> 4) & 63;
  const int b  = w >> 10;
  const int col  = lane & 15;
  const int half = lane >> 4;

  const bf16* Arow  = tfm + (size_t)(nt * 16 + col) * NLAT;      // A row m=col
  const bf16* Bslab = latT + ((size_t)b * 16 + j) * NLAT * 16;

  v8f acc = {};
#pragma unroll
  for (int c = 0; c < 4; ++c) {
    const int k0 = 32 * c;
    // A frag: lane half 0 -> K 0..7 & 16..23; half 1 -> K 8..15 & 24..31
    v8bf alo = *(const v8bf*)(Arow + k0 + 8 * half);
    v8bf ahi = *(const v8bf*)(Arow + k0 + 16 + 8 * half);
    v16bf a = __builtin_shufflevector(alo, ahi, 0, 1, 2, 3, 4, 5, 6, 7, 8, 9,
                                      10, 11, 12, 13, 14, 15);
    // B frag: lane holds col n(=i) = col; K in same table order as A
    v16bf bb;
#pragma unroll
    for (int s2 = 0; s2 < 16; ++s2) {
      const int l = k0 + (s2 < 8 ? 8 * half + s2 : 16 + 8 * half + (s2 - 8));
      bb[s2] = Bslab[(size_t)l * 16 + col];
    }
    acc = __builtin_amdgcn_wmma_f32_16x16x32_bf16(false, a, false, bb,
                                                  (short)0, acc, false, false);
  }
  // D: VGPR r -> n row (r or r+8 by lane half), col = lane%16 -> s
  const int s = j * 16 + col;
  float* obase = out + ((size_t)b * NIN + nt * 16) * STEPS + s;
#pragma unroll
  for (int r = 0; r < 8; ++r)
    __builtin_nontemporal_store(acc[r], &obase[(size_t)(r + 8 * half) * STEPS]);
}

// ------------------------------- launcher ----------------------------------
extern "C" void kernel_launch(void* const* d_in, const int* in_sizes, int n_in,
                              void* d_out, int out_size, void* d_ws,
                              size_t ws_size, hipStream_t stream) {
  (void)in_sizes; (void)n_in; (void)out_size; (void)ws_size;
  const float* x          = (const float*)d_in[0];   // [256][1024]
  const float* transform  = (const float*)d_in[1];   // [1024][128]
  const float* transition = (const float*)d_in[2];   // [128][128]
  // d_in[3] = steps (==256), baked into the decomposition.
  float* out = (float*)d_out;

  char* ws = (char*)d_ws;
  bf16*  tfm16 = (bf16*)(ws + 0);            // 262144 B
  float* lat0  = (float*)(ws + 262144);      // 131072 B
  float* T2    = (float*)(ws + 393216);      // 65536 B
  float* T4    = (float*)(ws + 458752);      // 65536 B
  float* T8    = (float*)(ws + 524288);      // 65536 B
  float* T16   = (float*)(ws + 589824);      // 65536 B
  float* R     = (float*)(ws + 655360);      // 2097152 B
  bf16*  latT  = (bf16*)(ws + 2752512);      // 16777216 B (~18.6 MB total)

  k_cvt_bf16<<<(NIN * NLAT + 255) / 256, 256, 0, stream>>>(transform, tfm16,
                                                           NIN * NLAT);
  // latent0 = x @ transform  (256x128, K=1024)
  k_gemm_f32<<<(BATCH * NLAT + 255) / 256, 256, 0, stream>>>(
      x, transform, lat0, BATCH, NLAT, NIN);
  // T^2, T^4, T^8, T^16 (fp32, 128^3 each)
  k_gemm_f32<<<64, 256, 0, stream>>>(transition, transition, T2, NLAT, NLAT, NLAT);
  k_gemm_f32<<<64, 256, 0, stream>>>(T2, T2, T4, NLAT, NLAT, NLAT);
  k_gemm_f32<<<64, 256, 0, stream>>>(T4, T4, T8, NLAT, NLAT, NLAT);
  k_gemm_f32<<<64, 256, 0, stream>>>(T8, T8, T16, NLAT, NLAT, NLAT);
  // R[j] = latent0 @ (T^16)^j, j = 0..15
  k_rchain<<<64, 256, 0, stream>>>(lat0, T16, R);
  // lats[16j+i] = R[j] @ T^(i+1) -> bf16 latT[b][j][l][i]
  k_lats<<<256, 256, 0, stream>>>(transition, R, latT);
  // out[b, n, 16j+i]
  k_proj<<<(BATCH * 64 * 16) / 8, 256, 0, stream>>>(tfm16, latT, out);
}
